// StackedUnidirLSTMDecoderwithEmbedding_23931557773553
// MI455X (gfx1250) — matrix operand order repaired and checked by reading, not compiled, not run on hardware
//
#include <hip/hip_runtime.h>
#include <math.h>

#define VOCAB  32000
#define EMB    512
#define HID    512
#define LAYERS 2
#define BATCH  32
#define SEQ    128
#define GATES  (4*HID)     // 2048
#define KCAT   (EMB+HID)   // 1024
#define KT_H   (HID/32)    // 16 K-tiles for a 512-wide operand

typedef __attribute__((ext_vector_type(16))) __bf16          v16bf;
typedef __attribute__((ext_vector_type(16))) unsigned short  v16us;
typedef __attribute__((ext_vector_type(8)))  float           v8f;
typedef __attribute__((ext_vector_type(8)))  unsigned int    v8u;

// float -> bf16, round-to-nearest-even
static __device__ __forceinline__ unsigned short f2bf(float f) {
  unsigned int u = __float_as_uint(f);
  u += 0x7fffu + ((u >> 16) & 1u);
  return (unsigned short)(u >> 16);
}

// Packed A-fragment index for element (m,n) of a 32x512 activation matrix.
// Layout: [mt][kt][lane][e], lane/e per ISA 7.12.2 16-bit A layout:
//   group g = lane>>4 holds K {8g..8g+7} U {16+8g..16+8g+7} of each 32-wide tile.
static __device__ __forceinline__ int packidx(int m, int n) {
  int kt   = n >> 5;
  int kl   = n & 31;
  int g    = (kl >> 3) & 1;
  int e    = (kl & 7) + 8 * (kl >> 4);
  int lane = (m & 15) + 16 * g;
  int mt   = m >> 4;
  return ((mt * KT_H + kt) * 32 + lane) * 16 + e;
}

// ---------------- one-time weight packing ----------------
// Packed B layout: flat = ((nt*KT + kt)*32 + lane)*16 + e
//   n = nt*16 + (lane&15);  k = kt*32 + 8*(lane>>4) + (e&7) + 16*(e>>3)   (B[k][n] = W[n][k])
__global__ void pack_generic(const float* __restrict__ W, unsigned short* __restrict__ out,
                             int N, int K) {
  const int KT = K / 32;
  long long flat = (long long)blockIdx.x * blockDim.x + threadIdx.x;
  long long total = (long long)N * K;
  if (flat >= total) return;
  int e  = (int)(flat & 15);
  int l  = (int)((flat >> 4) & 31);
  long long r = flat >> 9;
  int kt = (int)(r % KT);
  int nt = (int)(r / KT);
  int n  = nt * 16 + (l & 15);
  int k  = kt * 32 + 8 * (l >> 4) + (e & 7) + 16 * (e >> 3);
  out[flat] = f2bf(W[(long long)n * K + k]);
}

// Pack concatenated [W_ih ; W_hh] for one layer: K = 1024 (x-part then h-part)
__global__ void pack_wcat(const float* __restrict__ W_ih, const float* __restrict__ W_hh,
                          unsigned short* __restrict__ out) {
  const int KT = KCAT / 32;
  long long flat = (long long)blockIdx.x * blockDim.x + threadIdx.x;
  long long total = (long long)GATES * KCAT;
  if (flat >= total) return;
  int e  = (int)(flat & 15);
  int l  = (int)((flat >> 4) & 31);
  long long r = flat >> 9;
  int kt = (int)(r % KT);
  int nt = (int)(r / KT);
  int n  = nt * 16 + (l & 15);
  int k  = kt * 32 + 8 * (l >> 4) + (e & 7) + 16 * (e >> 3);
  float v = (k < EMB) ? W_ih[(long long)n * EMB + k]
                      : W_hh[(long long)n * HID + (k - EMB)];
  out[flat] = f2bf(v);
}

__global__ void init_state(const float* __restrict__ h0, const float* __restrict__ c0,
                           const float* __restrict__ b_ih, const float* __restrict__ b_hh,
                           float* __restrict__ h, float* __restrict__ c,
                           unsigned short* __restrict__ hp,   // packed bf16 h, per layer
                           float* __restrict__ bias) {
  int i = blockIdx.x * blockDim.x + threadIdx.x;
  if (i < LAYERS * BATCH * HID) {
    float hv = h0[i];
    h[i] = hv; c[i] = c0[i];
    int l = i >> 14;            // / (BATCH*HID)
    int rem = i & 16383;
    int m = rem >> 9, n = rem & 511;
    hp[l * (BATCH * HID) + packidx(m, n)] = f2bf(hv);
  }
  if (i < LAYERS * GATES) bias[i] = b_ih[i] + b_hh[i];
}

// ---------------- per-step kernels ----------------
// One block per batch row: argmax over VOCAB logits, then gather the embedding
// row directly into packed bf16 A-fragment layout.
__global__ void argmax_gather(const float* __restrict__ logits, long long row_stride,
                              const float* __restrict__ emb,
                              unsigned short* __restrict__ xp,
                              int* __restrict__ idx_out) {
  __shared__ float sval[256];
  __shared__ int   sidx[256];
  __shared__ int   s_best;
  int b = blockIdx.x;
  int t = threadIdx.x;
  const float* row = logits + (long long)b * row_stride;
  float best = -INFINITY; int bi = 0;
  for (int v = t; v < VOCAB; v += 256) {
    float x = row[v];
    if (x > best) { best = x; bi = v; }
  }
  sval[t] = best; sidx[t] = bi;
  __syncthreads();
  for (int s = 128; s > 0; s >>= 1) {
    if (t < s) {
      float ov = sval[t + s]; int oi = sidx[t + s];
      if (ov > sval[t] || (ov == sval[t] && oi < sidx[t])) { sval[t] = ov; sidx[t] = oi; }
    }
    __syncthreads();
  }
  if (t == 0) { s_best = sidx[0]; idx_out[b] = sidx[0]; }
  __syncthreads();
  int idx = s_best;
  for (int k = t; k < EMB; k += 256)
    xp[packidx(b, k)] = f2bf(emb[(long long)idx * EMB + k]);
}

// gates(32 x 2048) = [xin | hprev](32 x 1024) @ Wcat + bias ; bf16 WMMA, fp32 acc.
// A operands come pre-packed; K loop split into x-half and h-half so all A
// addressing is immediate-offset (no per-iteration cndmask pointer selects).
__global__ void __launch_bounds__(256)
lstm_gates(const unsigned short* __restrict__ ap0, const unsigned short* __restrict__ ap1,
           const unsigned short* __restrict__ Wp, const float* __restrict__ bias,
           float* __restrict__ gates) {
  const int KT = KCAT / 32;                                   // 32
  int wave = ((blockIdx.x * blockDim.x) + threadIdx.x) >> 5;  // 0..255
  int lane = threadIdx.x & 31;
  int mt = wave & 1;            // 2 M-tiles (batch 32)
  int nt = wave >> 1;           // 128 N-tiles (2048)
  int g  = lane >> 4;
  const v8u* bp = reinterpret_cast<const v8u*>(Wp) + ((long long)nt * KT) * 32 + lane;
  const v8u* a0 = reinterpret_cast<const v8u*>(ap0) + (long long)mt * KT_H * 32 + lane;
  const v8u* a1 = reinterpret_cast<const v8u*>(ap1) + (long long)mt * KT_H * 32 + lane;

  v8f acc = {};
#pragma unroll 4
  for (int kt = 0; kt < KT_H; ++kt) {           // x-part (K 0..511)
    v16bf bv = __builtin_bit_cast(v16bf, bp[kt * 32]);
    v16bf av = __builtin_bit_cast(v16bf, a0[kt * 32]);
    acc = __builtin_amdgcn_wmma_f32_16x16x32_bf16(false, av, false, bv,
                                                  (short)0, acc, false, false);
  }
#pragma unroll 4
  for (int kt = 0; kt < KT_H; ++kt) {           // h-part (K 512..1023)
    v16bf bv = __builtin_bit_cast(v16bf, bp[(KT_H + kt) * 32]);
    v16bf av = __builtin_bit_cast(v16bf, a1[kt * 32]);
    acc = __builtin_amdgcn_wmma_f32_16x16x32_bf16(false, av, false, bv,
                                                  (short)0, acc, false, false);
  }
  int n = nt * 16 + (lane & 15);
  float bb = bias[n];
#pragma unroll
  for (int v = 0; v < 8; ++v) {
    int mm = mt * 16 + v + 8 * g;                 // C/D layout: M = vgpr + 8*(lane>=16)
    gates[(long long)mm * GATES + n] = acc[v] + bb;
  }
}

// Elementwise cell update; also emits h_new in packed bf16 A-fragment layout.
__global__ void lstm_cell(const float* __restrict__ gates,
                          float* __restrict__ h, float* __restrict__ c,
                          unsigned short* __restrict__ hp) {
  int i = blockIdx.x * blockDim.x + threadIdx.x;
  if (i >= BATCH * HID) return;
  int m = i >> 9, n = i & 511;
  const float* gr = gates + (long long)m * GATES;
  float gi = gr[n], gf = gr[HID + n], gg = gr[2 * HID + n], go = gr[3 * HID + n];
  float si = 1.f / (1.f + __expf(-gi));
  float sf = 1.f / (1.f + __expf(-gf));
  float so = 1.f / (1.f + __expf(-go));
  float cn = sf * c[i] + si * tanhf(gg);
  float hn = so * tanhf(cn);
  c[i] = cn;
  h[i] = hn;
  hp[packidx(m, n)] = f2bf(hn);
}

// logits(32 x 32000) = h @ Wd^T + b ; 2 N-tiles per wave (A reuse, 2 indep. acc chains)
__global__ void __launch_bounds__(256)
dense_wmma(const unsigned short* __restrict__ hp1, const unsigned short* __restrict__ Wdp,
           const float* __restrict__ b_dense, float* __restrict__ out, int t) {
  const int KT = KT_H;                                        // 16
  int wave = ((blockIdx.x * blockDim.x) + threadIdx.x) >> 5;  // 0..1999 (exact grid)
  int lane = threadIdx.x & 31;
  int mt  = wave & 1;
  int nt0 = (wave >> 1) * 2;    // pair of N-tiles: nt0, nt0+1
  int g   = lane >> 4;
  const v8u* b0 = reinterpret_cast<const v8u*>(Wdp) + ((long long)nt0 * KT) * 32 + lane;
  const v8u* b1 = b0 + (long long)KT * 32;
  const v8u* ap = reinterpret_cast<const v8u*>(hp1) + (long long)mt * KT * 32 + lane;

  v8f acc0 = {}, acc1 = {};
#pragma unroll 8
  for (int kt = 0; kt < KT; ++kt) {
    __builtin_prefetch(b1 + (kt + 2) * 32, 0, 1);   // global_prefetch_b8 on B stream
    v16bf av  = __builtin_bit_cast(v16bf, ap[kt * 32]);
    v16bf bv0 = __builtin_bit_cast(v16bf, b0[kt * 32]);
    v16bf bv1 = __builtin_bit_cast(v16bf, b1[kt * 32]);
    acc0 = __builtin_amdgcn_wmma_f32_16x16x32_bf16(false, av, false, bv0,
                                                   (short)0, acc0, false, false);
    acc1 = __builtin_amdgcn_wmma_f32_16x16x32_bf16(false, av, false, bv1,
                                                   (short)0, acc1, false, false);
  }
  int n0 = nt0 * 16 + (lane & 15);
  float bb0 = b_dense[n0], bb1 = b_dense[n0 + 16];
#pragma unroll
  for (int v = 0; v < 8; ++v) {
    int mm = mt * 16 + v + 8 * g;
    long long base = ((long long)mm * SEQ + t) * VOCAB;
    out[base + n0]      = acc0[v] + bb0;
    out[base + n0 + 16] = acc1[v] + bb1;
  }
}

// ---------------- host ----------------
extern "C" void kernel_launch(void* const* d_in, const int* in_sizes, int n_in,
                              void* d_out, int out_size, void* d_ws, size_t ws_size,
                              hipStream_t stream) {
  const float* x       = (const float*)d_in[0];
  const float* h0      = (const float*)d_in[1];
  const float* c0      = (const float*)d_in[2];
  const float* emb     = (const float*)d_in[3];
  const float* W_ih    = (const float*)d_in[4];
  const float* W_hh    = (const float*)d_in[5];
  const float* b_ih    = (const float*)d_in[6];
  const float* b_hh    = (const float*)d_in[7];
  const float* W_dense = (const float*)d_in[8];
  const float* b_dense = (const float*)d_in[9];
  float* out = (float*)d_out;

  char* ws = (char*)d_ws;
  size_t off = 0;
  auto alloc = [&](size_t bytes) -> void* {
    off = (off + 255) & ~(size_t)255;
    void* p = ws + off;
    off += bytes;
    return p;
  };
  unsigned short* Wdp = (unsigned short*)alloc((size_t)VOCAB * HID * 2);           // 32.8 MB
  unsigned short* Wcp = (unsigned short*)alloc((size_t)LAYERS * GATES * KCAT * 2); // 8.4 MB
  float* bias = (float*)alloc((size_t)LAYERS * GATES * 4);
  float* hbuf = (float*)alloc((size_t)LAYERS * BATCH * HID * 4);
  float* cbuf = (float*)alloc((size_t)LAYERS * BATCH * HID * 4);
  unsigned short* xp = (unsigned short*)alloc((size_t)BATCH * EMB * 2);            // packed x
  unsigned short* hp = (unsigned short*)alloc((size_t)LAYERS * BATCH * HID * 2);   // packed h
  float* gbuf = (float*)alloc((size_t)BATCH * GATES * 4);
  int*   idx  = (int*)alloc((size_t)BATCH * 4);
  (void)ws_size; (void)in_sizes; (void)n_in; (void)out_size; (void)idx;

  // One-time (per call) packing + state init
  {
    long long totd = (long long)VOCAB * HID;
    pack_generic<<<(int)((totd + 255) / 256), 256, 0, stream>>>(W_dense, Wdp, VOCAB, HID);
    long long totc = (long long)GATES * KCAT;
    for (int l = 0; l < LAYERS; ++l)
      pack_wcat<<<(int)((totc + 255) / 256), 256, 0, stream>>>(
          W_ih + (long long)l * GATES * EMB, W_hh + (long long)l * GATES * HID,
          Wcp + (long long)l * GATES * KCAT);
    init_state<<<128, 256, 0, stream>>>(h0, c0, b_ih, b_hh, hbuf, cbuf, hp, bias);
  }

  const size_t HPL = (size_t)BATCH * HID;   // elements per layer in h buffers

  // Sequential decode scan
  for (int t = 0; t < SEQ; ++t) {
    const float* lsrc = (t == 0) ? x : (out + (long long)(t - 1) * VOCAB);
    long long lstride = (t == 0) ? (long long)VOCAB : (long long)SEQ * VOCAB;
    argmax_gather<<<BATCH, 256, 0, stream>>>(lsrc, lstride, emb, xp, idx);

    for (int l = 0; l < LAYERS; ++l) {
      const unsigned short* ain = (l == 0) ? xp : hp;     // layer1 input = layer0 packed h
      lstm_gates<<<32, 256, 0, stream>>>(ain, hp + (size_t)l * HPL,
                                         Wcp + (long long)l * GATES * KCAT,
                                         bias + (long long)l * GATES, gbuf);
      lstm_cell<<<64, 256, 0, stream>>>(gbuf, hbuf + (size_t)l * HPL,
                                        cbuf + (size_t)l * HPL, hp + (size_t)l * HPL);
    }
    dense_wmma<<<250, 256, 0, stream>>>(hp + HPL, Wdp, b_dense, out, t);
  }
}
